// KFLayer_14035953123566
// MI455X (gfx1250) — matrix operand order repaired and checked by reading, not compiled, not run on hardware
//
#include <hip/hip_runtime.h>
#include <math.h>
#include <stdint.h>

// ---------------------------------------------------------------------------
// Types for CDNA5 WMMA (wave32): V_WMMA_F32_16X16X32_BF16
// ---------------------------------------------------------------------------
typedef __bf16 bf16;
typedef bf16  v16bf __attribute__((ext_vector_type(16)));
typedef float v8f   __attribute__((ext_vector_type(8)));

#define EPSF 1e-7f

static constexpr int S_ROWS = 4096;   // queries
static constexpr int M_ROWS = 8192;   // keys / values
static constexpr int DK     = 256;    // spatial dim (time kept separately)
static constexpr int DO     = 256;

// ---------------------------------------------------------------------------
// CDNA5 async global->LDS copy (ASYNCcnt-tracked, no VGPR staging).
// LDS address = low 32 bits of the generic pointer (ISA 10.2 aperture rules).
// ---------------------------------------------------------------------------
__device__ __forceinline__ void async_copy16(uint32_t lds_off, const bf16* g) {
  asm volatile("global_load_async_to_lds_b128 %0, %1, off"
               :
               : "v"(lds_off), "v"(g)
               : "memory");
}
__device__ __forceinline__ void wait_async0() {
  asm volatile("s_wait_asynccnt 0x0" ::: "memory");
}

// ---------------------------------------------------------------------------
// Reductions (blockDim.x == 256 -> 8 wave32 waves)
// ---------------------------------------------------------------------------
__device__ __forceinline__ float warp_sum(float v) {
#pragma unroll
  for (int o = 16; o; o >>= 1) v += __shfl_xor(v, o, 32);
  return v;
}
__device__ __forceinline__ float warp_max(float v) {
#pragma unroll
  for (int o = 16; o; o >>= 1) v = fmaxf(v, __shfl_xor(v, o, 32));
  return v;
}
__device__ float block_sum(float v, float* sm) {
  int lane = threadIdx.x & 31, w = threadIdx.x >> 5;
  v = warp_sum(v);
  if (lane == 0) sm[w] = v;
  __syncthreads();
  if (w == 0) {
    float x = (lane < 8) ? sm[lane] : 0.0f;
    x = warp_sum(x);
    if (lane == 0) sm[0] = x;
  }
  __syncthreads();
  float r = sm[0];
  __syncthreads();
  return r;
}
__device__ float block_max(float v, float* sm) {
  int lane = threadIdx.x & 31, w = threadIdx.x >> 5;
  v = warp_max(v);
  if (lane == 0) sm[w] = v;
  __syncthreads();
  if (w == 0) {
    float x = (lane < 8) ? sm[lane] : -3.4e38f;
    x = warp_max(x);
    if (lane == 0) sm[0] = x;
  }
  __syncthreads();
  float r = sm[0];
  __syncthreads();
  return r;
}

// ---------------------------------------------------------------------------
// expmap0: row -> (time scalar f32, spatial bf16)
// ---------------------------------------------------------------------------
__global__ __launch_bounds__(256)
void expmap_kernel(const float* __restrict__ X, bf16* __restrict__ sp,
                   float* __restrict__ tm, int D) {
  __shared__ float sm[8];
  const int row = blockIdx.x;
  const float x = X[(size_t)row * D + threadIdx.x];
  const float n  = sqrtf(block_sum(x * x, sm));
  const float scale = fminf(3.0f / (n + 1e-7f), 1.0f);
  const float t = n * scale;                         // norm of tangent vector
  const float coef = sinhf(t) * scale / fmaxf(t, EPSF);
  sp[(size_t)row * D + threadIdx.x] = (bf16)(coef * x);
  if (threadIdx.x == 0) tm[row] = coshf(t);
}

// ---------------------------------------------------------------------------
// bf16 transpose: dst[c*rows + r] = src[r*cols + c]
// ---------------------------------------------------------------------------
__global__ void transpose_kernel(const bf16* __restrict__ src, bf16* __restrict__ dst,
                                 int rows, int cols) {
  size_t i = (size_t)blockIdx.x * blockDim.x + threadIdx.x;
  if (i >= (size_t)rows * cols) return;
  int r = (int)(i / cols), c = (int)(i % cols);
  dst[(size_t)c * rows + r] = src[i];
}

// ---------------------------------------------------------------------------
// z init: z = (1, 0...0); zero row accumulators
// ---------------------------------------------------------------------------
__global__ void init_z_kernel(float* zs, bf16* zsb, float* zt, float* srow,
                              float* emt, int S, int D) {
  size_t i = (size_t)blockIdx.x * blockDim.x + threadIdx.x;
  if (i < (size_t)S * D) { zs[i] = 0.0f; zsb[i] = (bf16)0.0f; }
  if (i < (size_t)S)     { zt[i] = 1.0f; srow[i] = 0.0f; emt[i] = 0.0f; }
}

// ---------------------------------------------------------------------------
// WMMA fragment loads from LDS per the CDNA5 16-bit layouts (ISA 7.12.2)
// A 16x32: lanes0-15 m, VGPR0-3 K=0..7, VGPR4-7 K=16..23; lanes16-31 +8
// B 32x16: lanes0-15 n, VGPR0-3 K=0..7, VGPR4-7 K=16..23; lanes16-31 +8
// ---------------------------------------------------------------------------
__device__ __forceinline__ v16bf frag_a(const bf16* __restrict__ row, int half) {
  v16bf f;
#pragma unroll
  for (int s = 0; s < 8; ++s) {
    int kb = (s < 4 ? 2 * s : 16 + 2 * (s - 4)) + half * 8;
    f[2 * s]     = row[kb];
    f[2 * s + 1] = row[kb + 1];
  }
  return f;
}
__device__ __forceinline__ v16bf frag_b(const bf16* __restrict__ row, int half) {
  v16bf f;
#pragma unroll
  for (int s = 0; s < 8; ++s) {
    int kb = (s < 4 ? 2 * s : 16 + 2 * (s - 4)) + half * 8;
    f[2 * s]     = row[kb];
    f[2 * s + 1] = row[kb + 1];
  }
  return f;
}

struct EpiArgs {
  const float* ta;     // time of A rows (modes 0,1)
  const float* tb;     // time of B rows (modes 0,1)
  const float* alpha;  // mode 1
  bf16*        cout;   // mode 1: c matrix (bf16)
  float*       srow;   // mode 1: sum_j c*inner   (atomic)
  float*       emt;    // mode 1: sum_j c*Vt[j]   (atomic)
  float*       outf;   // mode 0: sims; mode 2: euc_mean spatial
};

// ---------------------------------------------------------------------------
// C = A (MxK) * B(NxK)^T, bf16 in / f32 accumulate, fused epilogues.
// Block tile 128x128, 8 waves (2x4), wave tile 64x32 = 4x2 WMMA subtiles.
// Double-buffered LDS fed by global_load_async_to_lds_b128 (ASYNCcnt).
// MODE is a template arg so each instantiation has a minimal epilogue
// (avoids accumulator spills seen with a runtime-mode epilogue).
// ---------------------------------------------------------------------------
template <int MODE>
__global__ __launch_bounds__(256)
void gemm_nt_wmma_kernel(const bf16* __restrict__ A, const bf16* __restrict__ B,
                         int M, int N, int K, EpiArgs ea) {
  constexpr int BM = 128, BN = 128, BK = 32, LDT = 40;  // LDT pads bank conflicts
  __shared__ bf16 As[2 * BM * LDT];
  __shared__ bf16 Bs[2 * BN * LDT];

  const int tid  = threadIdx.x;
  const int lane = tid & 31;
  const int wid  = tid >> 5;
  const int wm   = wid & 1;    // wave row   (0..1) -> 64 rows each
  const int wn   = wid >> 1;   // wave col   (0..3) -> 32 cols each
  const int half = lane >> 4;
  const int l16  = lane & 15;
  const int rowA0 = blockIdx.y * BM;
  const int rowB0 = blockIdx.x * BN;

  // this thread's two 16B chunks per tile (A and B use the same pattern)
  const int r0 = tid >> 2;               // 0..63
  const int c0 = (tid & 3) << 3;         // 0,8,16,24
  const int r1 = r0 + 64;

  v8f acc[4][2];
#pragma unroll
  for (int i = 0; i < 4; ++i)
#pragma unroll
    for (int j = 0; j < 2; ++j)
#pragma unroll
      for (int e = 0; e < 8; ++e) acc[i][j][e] = 0.0f;

  const int nk = K / BK;

  auto issue_tile = [&](int kt) {
    const int k0  = kt * BK;
    bf16* Ad = As + (kt & 1) * (BM * LDT);
    bf16* Bd = Bs + (kt & 1) * (BN * LDT);
    async_copy16((uint32_t)(uintptr_t)&Ad[r0 * LDT + c0],
                 &A[(size_t)(rowA0 + r0) * K + k0 + c0]);
    async_copy16((uint32_t)(uintptr_t)&Ad[r1 * LDT + c0],
                 &A[(size_t)(rowA0 + r1) * K + k0 + c0]);
    async_copy16((uint32_t)(uintptr_t)&Bd[r0 * LDT + c0],
                 &B[(size_t)(rowB0 + r0) * K + k0 + c0]);
    async_copy16((uint32_t)(uintptr_t)&Bd[r1 * LDT + c0],
                 &B[(size_t)(rowB0 + r1) * K + k0 + c0]);
  };

  issue_tile(0);
  wait_async0();
  __syncthreads();

  for (int kt = 0; kt < nk; ++kt) {
    const bf16* Ac = As + (kt & 1) * (BM * LDT);
    const bf16* Bc = Bs + (kt & 1) * (BN * LDT);

    if (kt + 1 < nk) issue_tile(kt + 1);   // overlap with WMMA below

    v16bf af[4], bfv[2];
#pragma unroll
    for (int mi = 0; mi < 4; ++mi)
      af[mi] = frag_a(&Ac[(wm * 64 + mi * 16 + l16) * LDT], half);
#pragma unroll
    for (int ni = 0; ni < 2; ++ni)
      bfv[ni] = frag_b(&Bc[(wn * 32 + ni * 16 + l16) * LDT], half);

#pragma unroll
    for (int mi = 0; mi < 4; ++mi)
#pragma unroll
      for (int ni = 0; ni < 2; ++ni)
        acc[mi][ni] = __builtin_amdgcn_wmma_f32_16x16x32_bf16(
            false, af[mi], false, bfv[ni], (short)0, acc[mi][ni], false, false);

    wait_async0();      // next tile resident in the other buffer
    __syncthreads();    // all waves done reading current buffer
  }

  // Epilogue. C/D layout: VGPR v -> m_local = v + 8*half, n_local = l16.
#pragma unroll
  for (int mi = 0; mi < 4; ++mi) {
#pragma unroll
    for (int ni = 0; ni < 2; ++ni) {
      const int n = rowB0 + wn * 32 + ni * 16 + l16;
      const float tb_n = (MODE <= 1) ? ea.tb[n] : 0.0f;
#pragma unroll
      for (int v = 0; v < 8; ++v) {
        const int m = rowA0 + wm * 64 + mi * 16 + v + half * 8;
        const float a = acc[mi][ni][v];
        if (MODE == 0) {
          ea.outf[(size_t)m * N + n] = a - ea.ta[m] * tb_n;      // sims
        } else if (MODE == 1) {
          const float inner = a - ea.ta[m] * tb_n;               // <z,v>_L
          const float al    = ea.alpha[(size_t)m * N + n];
          const float dist  = acoshf(fmaxf(-inner, 1.0f + EPSF));
          const float den   = sqrtf(fmaxf(inner * inner - 1.0f, EPSF));
          const float c     = al * dist / den;
          ea.cout[(size_t)m * N + n] = (bf16)c;
          float vs = c * inner;
          float vt = c * tb_n;
#pragma unroll
          for (int o = 1; o < 16; o <<= 1) {                     // reduce 16 n's
            vs += __shfl_xor(vs, o, 32);
            vt += __shfl_xor(vt, o, 32);
          }
          if (l16 == 0) {
            atomicAdd(&ea.srow[m], vs);
            atomicAdd(&ea.emt[m], vt);
          }
        } else {
          ea.outf[(size_t)m * N + n] = a;                        // euc_mean_sp
        }
      }
    }
  }
}

// ---------------------------------------------------------------------------
// softmax(-beta * sims) over rows of 8192, in place
// ---------------------------------------------------------------------------
__global__ __launch_bounds__(256)
void softmax_neg_kernel(float* __restrict__ a, int N, float beta) {
  __shared__ float sm[8];
  float* p = a + (size_t)blockIdx.x * N;
  float v[32];
  float mx = -3.4e38f;
#pragma unroll
  for (int k = 0; k < 32; ++k) {
    v[k] = -beta * p[threadIdx.x + (k << 8)];
    mx = fmaxf(mx, v[k]);
  }
  mx = block_max(mx, sm);
  float s = 0.0f;
#pragma unroll
  for (int k = 0; k < 32; ++k) { v[k] = __expf(v[k] - mx); s += v[k]; }
  s = block_sum(s, sm);
  const float inv = 1.0f / s;
#pragma unroll
  for (int k = 0; k < 32; ++k) p[threadIdx.x + (k << 8)] = v[k] * inv;
}

// ---------------------------------------------------------------------------
// Karcher z-update (block per row); also re-zeroes row accumulators
// ---------------------------------------------------------------------------
__global__ __launch_bounds__(256)
void z_update_kernel(float* __restrict__ zs, bf16* __restrict__ zsb,
                     float* __restrict__ zt, const float* __restrict__ em,
                     float* __restrict__ srow, float* __restrict__ emt, int D) {
  __shared__ float sm[8];
  const int row = blockIdx.x;
  const size_t base = (size_t)row * D + threadIdx.x;
  const float s   = srow[row];
  const float zt0 = zt[row];
  const float z0  = zs[base];
  const float e   = em[base] + s * z0;           // euc_mean spatial
  const float et  = emt[row] + s * zt0;          // euc_mean time
  const float un2 = block_sum(e * e, sm) - et * et;
  const float un  = sqrtf(fmaxf(un2, EPSF));
  const float ch  = coshf(un);
  const float sh  = sinhf(un) / un;
  const float zn  = ch * z0 + sh * e;
  zs[base]  = zn;
  zsb[base] = (bf16)zn;
  if (threadIdx.x == 0) {
    zt[row]   = ch * zt0 + sh * et;
    srow[row] = 0.0f;
    emt[row]  = 0.0f;
  }
}

// ---------------------------------------------------------------------------
// logmap0 spatial -> output
// ---------------------------------------------------------------------------
__global__ __launch_bounds__(256)
void logmap_kernel(const float* __restrict__ zs, const float* __restrict__ zt,
                   float* __restrict__ out, int D) {
  __shared__ float sm[8];
  const int row = blockIdx.x;
  const float z = zs[(size_t)row * D + threadIdx.x];
  const float n = sqrtf(block_sum(z * z, sm));
  const float dist = acoshf(fmaxf(zt[row], 1.0f + EPSF));
  out[(size_t)row * D + threadIdx.x] = dist * z / fmaxf(n, EPSF);
}

// ---------------------------------------------------------------------------
extern "C" void kernel_launch(void* const* d_in, const int* in_sizes, int n_in,
                              void* d_out, int out_size, void* d_ws, size_t ws_size,
                              hipStream_t stream) {
  (void)in_sizes; (void)n_in; (void)out_size; (void)ws_size;
  const float* R  = (const float*)d_in[0];
  const float* WK = (const float*)d_in[1];
  const float* WV = (const float*)d_in[2];
  const int steps = 3;  // setup_inputs: karcher_steps = 3 (device scalar, not host-readable)

  char* w = (char*)d_ws;
  auto alloc = [&](size_t bytes) -> char* {
    char* p = w;
    w += (bytes + 255) & ~(size_t)255;
    return p;
  };
  float* alpha = (float*)alloc((size_t)S_ROWS * M_ROWS * 4);   // sims -> alpha
  bf16*  cmat  = (bf16*) alloc((size_t)S_ROWS * M_ROWS * 2);
  bf16*  Rsp   = (bf16*) alloc((size_t)S_ROWS * DK * 2);
  bf16*  Ksp   = (bf16*) alloc((size_t)M_ROWS * DK * 2);
  bf16*  Vsp   = (bf16*) alloc((size_t)M_ROWS * DO * 2);
  bf16*  VspT  = (bf16*) alloc((size_t)M_ROWS * DO * 2);
  float* Rt    = (float*)alloc((size_t)S_ROWS * 4);
  float* Kt    = (float*)alloc((size_t)M_ROWS * 4);
  float* Vt    = (float*)alloc((size_t)M_ROWS * 4);
  float* zs    = (float*)alloc((size_t)S_ROWS * DO * 4);
  bf16*  zsb   = (bf16*) alloc((size_t)S_ROWS * DO * 2);
  float* ztim  = (float*)alloc((size_t)S_ROWS * 4);
  float* em    = (float*)alloc((size_t)S_ROWS * DO * 4);
  float* srow  = (float*)alloc((size_t)S_ROWS * 4);
  float* emt   = (float*)alloc((size_t)S_ROWS * 4);

  expmap_kernel<<<S_ROWS, 256, 0, stream>>>(R,  Rsp, Rt, DK);
  expmap_kernel<<<M_ROWS, 256, 0, stream>>>(WK, Ksp, Kt, DK);
  expmap_kernel<<<M_ROWS, 256, 0, stream>>>(WV, Vsp, Vt, DO);

  {
    size_t tot = (size_t)M_ROWS * DO;
    transpose_kernel<<<(int)((tot + 255) / 256), 256, 0, stream>>>(Vsp, VspT, M_ROWS, DO);
  }
  init_z_kernel<<<(S_ROWS * DO + 255) / 256, 256, 0, stream>>>(zs, zsb, ztim, srow, emt,
                                                               S_ROWS, DO);

  // sims = Rsp @ Ksp^T - Rt (x) Kt
  EpiArgs e0{Rt, Kt, nullptr, nullptr, nullptr, nullptr, alpha};
  gemm_nt_wmma_kernel<0><<<dim3(M_ROWS / 128, S_ROWS / 128), 256, 0, stream>>>(
      Rsp, Ksp, S_ROWS, M_ROWS, DK, e0);
  softmax_neg_kernel<<<S_ROWS, 256, 0, stream>>>(alpha, M_ROWS, 1.0f / 16.0f);

  for (int it = 0; it < steps; ++it) {
    // c = alpha * dist / denom from inner_zv computed in-register
    EpiArgs e1{ztim, Vt, alpha, cmat, srow, emt, nullptr};
    gemm_nt_wmma_kernel<1><<<dim3(M_ROWS / 128, S_ROWS / 128), 256, 0, stream>>>(
        zsb, Vsp, S_ROWS, M_ROWS, DO, e1);
    // euc_mean spatial = c @ Vsp   (via VspT as NT)
    EpiArgs e2{nullptr, nullptr, nullptr, nullptr, nullptr, nullptr, em};
    gemm_nt_wmma_kernel<2><<<dim3(DO / 128, S_ROWS / 128), 256, 0, stream>>>(
        cmat, VspT, S_ROWS, DO, M_ROWS, e2);
    z_update_kernel<<<S_ROWS, 256, 0, stream>>>(zs, zsb, ztim, em, srow, emt, DO);
  }

  logmap_kernel<<<S_ROWS, 256, 0, stream>>>(zs, ztim, (float*)d_out, DO);
}
